// WindowAttention_21440476741823
// MI455X (gfx1250) — matrix operand order repaired and checked by reading, not compiled, not run on hardware
//
#include <hip/hip_runtime.h>

typedef __bf16 v16bf  __attribute__((ext_vector_type(16)));
typedef float  v8f    __attribute__((ext_vector_type(8)));
typedef __bf16 bf16x2 __attribute__((ext_vector_type(2)));

union AB { unsigned u[8]; v16bf v; };

#define N_TOK 343
#define NPAD  352          /* 22 * 16 */
#define NN    117649       /* 343*343 */
#define SCALE 0.17677669529663687f   /* 32^-0.5 */

// ---------------------------------------------------------------------------
// Kernel A: fp32 -> bf16 bulk convert (2 elements / thread)
// ---------------------------------------------------------------------------
__global__ void k_cvt(const float* __restrict__ in, __bf16* __restrict__ out, int n) {
    int i = (blockIdx.x * 256 + threadIdx.x) * 2;
    if (i >= n) return;
    float2 v = *(const float2*)(in + i);
    bf16x2 o; o.x = (__bf16)v.x; o.y = (__bf16)v.y;
    *(bf16x2*)(out + i) = o;
}

// ---------------------------------------------------------------------------
// Kernel B: zero pad rows/cols of Q, K (rows 343..351) and Vt (keys 343..351)
// ---------------------------------------------------------------------------
__global__ void k_padzero(__bf16* __restrict__ Q, __bf16* __restrict__ K,
                          __bf16* __restrict__ Vt) {
    int idx = blockIdx.x * 256 + threadIdx.x;
    if (idx >= 1536 * 288) return;
    int bh = idx / 288, r = idx % 288;
    int row = 343 + (r >> 5), d = r & 31;
    Q [(bh * 352 + row) * 32 + d] = (__bf16)0.f;
    K [(bh * 352 + row) * 32 + d] = (__bf16)0.f;
    Vt[(bh * 32 + d) * 352 + row] = (__bf16)0.f;
}

// ---------------------------------------------------------------------------
// Kernel C: expand relative-position bias table -> bias_f[h][i][j]
// ---------------------------------------------------------------------------
__global__ void k_bias(const int* __restrict__ rel, const float* __restrict__ table,
                       float* __restrict__ biasF) {
    int idx = blockIdx.x * 256 + threadIdx.x;
    if (idx >= NN) return;
    int r = rel[idx];
    #pragma unroll
    for (int h = 0; h < 6; ++h)
        biasF[h * NN + idx] = table[r * 6 + h];
}

// ---------------------------------------------------------------------------
// Kernel 1: qkv = xb @ wb^T + qkv_b -> Q(scaled)/K bf16 [bh,352,32], Vt [bh,32,352]
// bias pre-seeded into the WMMA C accumulator
// ---------------------------------------------------------------------------
__global__ __launch_bounds__(32) void k_qkv(const __bf16* __restrict__ xb,
                                            const __bf16* __restrict__ wb,
                                            const float* __restrict__ bias,
                                            __bf16* __restrict__ Q,
                                            __bf16* __restrict__ K,
                                            __bf16* __restrict__ Vt) {
    const int nb = blockIdx.x * 16;      // column base (0..575)
    const int mb = blockIdx.y * 16;      // row base (0..87807)
    const int lane = threadIdx.x, half = lane >> 4, l15 = lane & 15;
    const int rowA = mb + l15;
    const int colB = nb + l15;
    const float bv = bias[colB];
    v8f acc;                             // C seeded with per-column bias
    #pragma unroll
    for (int j = 0; j < 8; ++j) acc[j] = bv;
    #pragma unroll
    for (int kk = 0; kk < 192; kk += 32) {
        AB a, b;
        #pragma unroll
        for (int j = 0; j < 8; ++j) {
            int ka = kk + (j < 4 ? 0 : 16) + (j & 3) * 2 + half * 8;
            a.u[j] = *(const unsigned*)(xb + rowA * 192 + ka);
            int kb = kk + (j < 4 ? 0 : 16) + (j & 3) * 2 + half * 8;
            b.u[j] = *(const unsigned*)(wb + colB * 192 + kb);
        }
        acc = __builtin_amdgcn_wmma_f32_16x16x32_bf16(false, a.v, false, b.v,
                                                      (short)0, acc, false, false);
    }
    const int s   = nb / 192;                 // qkv split: uniform per tile
    const int nbm = nb - s * 192;
    const int h   = (nbm + l15) >> 5;
    const int d   = (nbm + l15) & 31;
    const int bw0 = mb / 343;                 // one scalar division
    const int r0  = mb - bw0 * 343;
    if (s == 2) {                             // V: transposed, padded
        #pragma unroll
        for (int j = 0; j < 8; ++j) {
            int nq = r0 + j + half * 8, bw = bw0;
            if (nq >= 343) { nq -= 343; ++bw; }
            Vt[((bw * 6 + h) * 32 + d) * 352 + nq] = (__bf16)acc[j];
        }
    } else {                                  // Q (scaled) / K: row-major, padded
        __bf16* dst = s ? K : Q;
        const float mul = s ? 1.0f : SCALE;
        #pragma unroll
        for (int j = 0; j < 8; ++j) {
            int nq = r0 + j + half * 8, bw = bw0;
            if (nq >= 343) { nq -= 343; ++bw; }
            dst[((bw * 6 + h) * 352 + nq) * 32 + d] = (__bf16)(acc[j] * mul);
        }
    }
}

// ---------------------------------------------------------------------------
// Kernel 2: fused attention. One wave per (b, h, 16-row block).
// bias+mask enter through the WMMA C operand; S band (16x352) in 22 f32
// accumulators; softmax in-register; P via LDS (C/D -> A layout), then P@V.
// ---------------------------------------------------------------------------
__global__ __launch_bounds__(32) void k_attn(const __bf16* __restrict__ Q,
                                             const __bf16* __restrict__ Kb,
                                             const __bf16* __restrict__ Vt,
                                             const float* __restrict__ biasF,
                                             const float* __restrict__ mask,
                                             __bf16* __restrict__ ctx) {
    __shared__ __bf16 P[16 * NPAD];
    const int rblk = blockIdx.x, h = blockIdx.y, b = blockIdx.z;
    const int w = b & 63;                       // window id = b % 64
    const int lane = threadIdx.x, half = lane >> 4, l15 = lane & 15;
    const int bh = b * 6 + h;

    // --- A operand: 16 query rows (padded buffer: no clamp needed) ----------
    AB a;
    const int rowA = rblk * 16 + l15;
    #pragma unroll
    for (int j = 0; j < 8; ++j) {
        int d = (j < 4 ? 0 : 16) + (j & 3) * 2 + half * 8;
        a.u[j] = *(const unsigned*)(Q + (bh * 352 + rowA) * 32 + d);
    }

    // --- S = q k^T + (bias + mask) via WMMA C, 22 column tiles --------------
    v8f S[22];
    #pragma unroll
    for (int t = 0; t < 22; ++t) {
        AB bm;
        const int key = t * 16 + l15;
        #pragma unroll
        for (int v = 0; v < 8; ++v) {
            int d = (v < 4 ? 0 : 16) + (v & 3) * 2 + half * 8;
            bm.u[v] = *(const unsigned*)(Kb + (bh * 352 + key) * 32 + d);
        }
        v8f cin = {};
        #pragma unroll
        for (int j = 0; j < 8; ++j) {
            int i = rblk * 16 + j + half * 8;
            if (key < 343 && i < 343)
                cin[j] = biasF[h * NN + i * 343 + key] + mask[w * NN + i * 343 + key];
        }
        S[t] = __builtin_amdgcn_wmma_f32_16x16x32_bf16(false, a.v, false, bm.v,
                                                       (short)0, cin, false, false);
        #pragma unroll
        for (int j = 0; j < 8; ++j)
            if (key >= 343) S[t][j] = -1e30f;   // pad columns -> -inf (t==21 only)
    }

    // --- row softmax (rows split across 16-lane halves) ---------------------
    float mrow[8], rinv[8];
    #pragma unroll
    for (int j = 0; j < 8; ++j) {
        float m = -1e30f;
        #pragma unroll
        for (int t = 0; t < 22; ++t) m = fmaxf(m, S[t][j]);
        #pragma unroll
        for (int o = 8; o >= 1; o >>= 1) m = fmaxf(m, __shfl_xor(m, o, 32));
        mrow[j] = m;
    }
    #pragma unroll
    for (int j = 0; j < 8; ++j) {
        float s = 0.f;
        #pragma unroll
        for (int t = 0; t < 22; ++t) { float e = __expf(S[t][j] - mrow[j]); S[t][j] = e; s += e; }
        #pragma unroll
        for (int o = 8; o >= 1; o >>= 1) s += __shfl_xor(s, o, 32);
        rinv[j] = 1.0f / s;
    }

    // --- P -> LDS (bf16, row-major 16 x 352) --------------------------------
    #pragma unroll
    for (int t = 0; t < 22; ++t) {
        #pragma unroll
        for (int j = 0; j < 8; ++j)
            P[(j + half * 8) * NPAD + t * 16 + l15] = (__bf16)(S[t][j] * rinv[j]);
    }
    __syncthreads();

    // --- O = P @ V : 11 K-chunks x 2 head-dim tiles, all contiguous loads ---
    v8f O0 = {}, O1 = {};
    const int vbase0 = (bh * 32 + l15) * 352;        // n = l15
    const int vbase1 = (bh * 32 + 16 + l15) * 352;   // n = 16 + l15
    #pragma unroll
    for (int c = 0; c < 11; ++c) {
        AB pa, b0, b1;
        #pragma unroll
        for (int j = 0; j < 8; ++j) {
            int k = c * 32 + (j < 4 ? 0 : 16) + (j & 3) * 2 + half * 8;
            pa.u[j] = *(const unsigned*)(&P[l15 * NPAD + k]);
        }
        #pragma unroll
        for (int v = 0; v < 8; ++v) {
            int key = c * 32 + (v < 4 ? 0 : 16) + (v & 3) * 2 + half * 8;
            b0.u[v] = *(const unsigned*)(Vt + vbase0 + key);
            b1.u[v] = *(const unsigned*)(Vt + vbase1 + key);
        }
        O0 = __builtin_amdgcn_wmma_f32_16x16x32_bf16(false, pa.v, false, b0.v,
                                                     (short)0, O0, false, false);
        O1 = __builtin_amdgcn_wmma_f32_16x16x32_bf16(false, pa.v, false, b1.v,
                                                     (short)0, O1, false, false);
    }
    #pragma unroll
    for (int j = 0; j < 8; ++j) {
        int row = rblk * 16 + j + half * 8;
        if (row < 343) {
            ctx[(b * 343 + row) * 192 + h * 32 + l15]      = (__bf16)O0[j];
            ctx[(b * 343 + row) * 192 + h * 32 + 16 + l15] = (__bf16)O1[j];
        }
    }
}

// ---------------------------------------------------------------------------
// Kernel 3: out = ctx @ proj_w^T + proj_b  (fp32 out), bias seeded into C
// ---------------------------------------------------------------------------
__global__ __launch_bounds__(32) void k_proj(const __bf16* __restrict__ ctx,
                                             const __bf16* __restrict__ wb,
                                             const float* __restrict__ bias,
                                             float* __restrict__ out) {
    const int nb = blockIdx.x * 16, mb = blockIdx.y * 16;
    const int lane = threadIdx.x, half = lane >> 4, l15 = lane & 15;
    const int rowA = mb + l15, colB = nb + l15;
    const float bv = bias[colB];
    v8f acc;
    #pragma unroll
    for (int j = 0; j < 8; ++j) acc[j] = bv;
    #pragma unroll
    for (int kk = 0; kk < 192; kk += 32) {
        AB a, b;
        #pragma unroll
        for (int j = 0; j < 8; ++j) {
            int ka = kk + (j < 4 ? 0 : 16) + (j & 3) * 2 + half * 8;
            a.u[j] = *(const unsigned*)(ctx + rowA * 192 + ka);
            int kb = kk + (j < 4 ? 0 : 16) + (j & 3) * 2 + half * 8;
            b.u[j] = *(const unsigned*)(wb + colB * 192 + kb);
        }
        acc = __builtin_amdgcn_wmma_f32_16x16x32_bf16(false, a.v, false, b.v,
                                                      (short)0, acc, false, false);
    }
    #pragma unroll
    for (int j = 0; j < 8; ++j) {
        int row = mb + j + half * 8;
        out[row * 192 + colB] = acc[j];
    }
}

// ---------------------------------------------------------------------------
extern "C" void kernel_launch(void* const* d_in, const int* in_sizes, int n_in,
                              void* d_out, int out_size, void* d_ws, size_t ws_size,
                              hipStream_t stream) {
    const float* x      = (const float*)d_in[0];
    const float* mask   = (const float*)d_in[1];
    const float* qkv_w  = (const float*)d_in[2];
    const float* qkv_b  = (const float*)d_in[3];
    const float* proj_w = (const float*)d_in[4];
    const float* proj_b = (const float*)d_in[5];
    const float* btable = (const float*)d_in[6];
    const int*   rel    = (const int*)d_in[7];

    // workspace layout (bytes):
    //   Q | K | Vt      : bf16, 34,603,008 B each (padded 352)
    //   xb (== ctx)     : bf16, 33,718,272 B   (ctx aliases xb: xb dead after k_qkv)
    //   bias_f          : f32,   2,823,576 B
    //   wqkv | wproj    : bf16,    221,184 | 73,728 B
    const size_t QKB = (size_t)1536 * 352 * 32 * 2;   // 34,603,008
    char* ws = (char*)d_ws;
    __bf16* Q     = (__bf16*)(ws);
    __bf16* K     = (__bf16*)(ws + 1 * QKB);
    __bf16* Vt    = (__bf16*)(ws + 2 * QKB);
    __bf16* xb    = (__bf16*)(ws + 3 * QKB);
    __bf16* ctx   = xb;                                // alias (serialized on stream)
    float*  bf    = (float*) (ws + 3 * QKB + 33718272);
    __bf16* wqkv  = (__bf16*)(ws + 3 * QKB + 33718272 + 2823576);
    __bf16* wproj = (__bf16*)(ws + 3 * QKB + 33718272 + 2823576 + 221184);

    const int XN = 256 * 343 * 192;      // 16,859,136
    const int WQ = 576 * 192;            // 110,592
    const int WP = 192 * 192;            // 36,864
    k_cvt<<<(XN / 2 + 255) / 256, 256, 0, stream>>>(x, xb, XN);
    k_cvt<<<(WQ / 2 + 255) / 256, 256, 0, stream>>>(qkv_w, wqkv, WQ);
    k_cvt<<<(WP / 2 + 255) / 256, 256, 0, stream>>>(proj_w, wproj, WP);
    k_padzero<<<(1536 * 288 + 255) / 256, 256, 0, stream>>>(Q, K, Vt);
    k_bias<<<(NN + 255) / 256, 256, 0, stream>>>(rel, btable, bf);

    k_qkv <<<dim3(36, 5488), 32, 0, stream>>>(xb, wqkv, qkv_b, Q, K, Vt);
    k_attn<<<dim3(22, 6, 256), 32, 0, stream>>>(Q, K, Vt, bf, mask, ctx);
    k_proj<<<dim3(12, 5488), 32, 0, stream>>>(ctx, wproj, proj_b, (float*)d_out);
}
